// FullGapModel_36584531427348
// MI455X (gfx1250) — matrix operand order repaired and checked by reading, not compiled, not run on hardware
//
#include <hip/hip_runtime.h>

typedef __attribute__((ext_vector_type(16))) _Float16 v16h;
typedef __attribute__((ext_vector_type(8)))  _Float16 v8h;
typedef __attribute__((ext_vector_type(8)))  float    v8f;

#define DD 128      // feature dim (fixed by the reference)
#define WAVES 8     // waves per block (wave32) -> 128 atoms / block

// ---------------------------------------------------------------------------
// Phase 0: wm[m] = weights[0, support_segments[m]]   (P == 1)
// ---------------------------------------------------------------------------
__global__ void build_wm_kernel(const float* __restrict__ weights,
                                const int* __restrict__ supseg,
                                float* __restrict__ wm, int M) {
  int m = blockIdx.x * blockDim.x + threadIdx.x;
  if (m < M) wm[m] = weights[supseg[m]];
}

// ---------------------------------------------------------------------------
// Phase 1: A[i][j] = sum_m wm[m] * sp[m][i] * sp[m][j]   (f32 accum -> f16)
// Grid: 8x8 = 64 blocks of 256 threads; each block owns a 16x16 (i,j) tile.
// ---------------------------------------------------------------------------
__global__ __launch_bounds__(256)
void build_amat_kernel(const float* __restrict__ sp,
                       const float* __restrict__ wm,
                       _Float16* __restrict__ Amat, int M) {
  __shared__ float sA[64][16];   // wm[m]*sp[m][i-tile]
  __shared__ float sB[64][16];   // sp[m][j-tile]
  const int tid = threadIdx.x;
  const int ti  = tid & 15;
  const int tj  = tid >> 4;
  const int bi  = (blockIdx.x & 7) * 16;
  const int bj  = (blockIdx.x >> 3) * 16;
  float acc = 0.f;
  for (int m0 = 0; m0 < M; m0 += 64) {
    __syncthreads();
    for (int t = tid; t < 64 * 16; t += 256) {
      int mm = t >> 4, cc = t & 15;
      int m = m0 + mm;
      float a = 0.f, b = 0.f;
      if (m < M) {
        float w = wm[m];
        a = w * sp[(size_t)m * DD + bi + cc];
        b =     sp[(size_t)m * DD + bj + cc];
      }
      sA[mm][cc] = a;
      sB[mm][cc] = b;
    }
    __syncthreads();
#pragma unroll 16
    for (int mm = 0; mm < 64; ++mm) acc += sA[mm][ti] * sB[mm][tj];
  }
  Amat[(size_t)(bi + ti) * DD + (bj + tj)] = (_Float16)acc;
}

// ---------------------------------------------------------------------------
// Phase 2: per-atom quadratic form q_n = x̂ᵀ A x̂ via WMMA, atomically
// accumulated into out[atom_segments[n]].
// ---------------------------------------------------------------------------
__global__ __launch_bounds__(32 * WAVES)
void gap_quadform_kernel(const float* __restrict__ ps,
                         const _Float16* __restrict__ Amat,
                         const int* __restrict__ atom_seg,
                         float* __restrict__ out, int N) {
  __shared__ __align__(16) _Float16 sAmat[DD * DD];          // 32 KB, block-shared
  __shared__ __align__(16) _Float16 sXhat[WAVES][16 * DD];   // row-major x̂ per wave
  __shared__ __align__(16) _Float16 sXhatT[WAVES][DD * 16];  // transposed x̂ per wave

  const int tid  = threadIdx.x;
  const int lane = tid & 31;
  const int w    = tid >> 5;
  const int m16  = lane & 15;
  const int half = lane >> 4;   // 0 or 1

  // Cooperative load of A into LDS (2048 x uint4, 8 per thread)
  {
    const uint4* src = (const uint4*)Amat;
    uint4*       dst = (uint4*)sAmat;
#pragma unroll
    for (int i = 0; i < (DD * DD / 8) / (32 * WAVES); ++i)
      dst[tid + i * 32 * WAVES] = src[tid + i * 32 * WAVES];
  }

  const int atomBase = (blockIdx.x * WAVES + w) * 16;

  // Load 16 rows, compute inv-norm wave-wide, write f16 x̂ to LDS (both layouts)
  for (int i = 0; i < 16; ++i) {
    int atom = atomBase + i;
    float4 v = make_float4(0.f, 0.f, 0.f, 0.f);
    if (atom < N) v = ((const float4*)ps)[(size_t)atom * (DD / 4) + lane];
    float s = v.x * v.x + v.y * v.y + v.z * v.z + v.w * v.w;
    s += __shfl_xor(s, 1, 32);
    s += __shfl_xor(s, 2, 32);
    s += __shfl_xor(s, 4, 32);
    s += __shfl_xor(s, 8, 32);
    s += __shfl_xor(s, 16, 32);
    float invn = (s > 0.f) ? rsqrtf(s) : 0.f;
    _Float16 h0 = (_Float16)(v.x * invn);
    _Float16 h1 = (_Float16)(v.y * invn);
    _Float16 h2 = (_Float16)(v.z * invn);
    _Float16 h3 = (_Float16)(v.w * invn);
    const int c0 = lane * 4;
    _Float16* rowp = &sXhat[w][i * DD + c0];
    rowp[0] = h0; rowp[1] = h1; rowp[2] = h2; rowp[3] = h3;
    sXhatT[w][(c0 + 0) * 16 + i] = h0;
    sXhatT[w][(c0 + 1) * 16 + i] = h1;
    sXhatT[w][(c0 + 2) * 16 + i] = h2;
    sXhatT[w][(c0 + 3) * 16 + i] = h3;
  }
  __syncthreads();

  // A-operand (16x32 f16): lanes 0-15 hold M=lane, K {c*32+0..7, c*32+16..23};
  // lanes 16-31 hold K {c*32+8..15, c*32+24..31}.
  v16h a[4];
#pragma unroll
  for (int c = 0; c < 4; ++c) {
    const int kb = c * 32 + half * 8;
    v8h lo = *(const v8h*)&sXhat[w][m16 * DD + kb];
    v8h hi = *(const v8h*)&sXhat[w][m16 * DD + kb + 16];
#pragma unroll
    for (int h = 0; h < 8; ++h) { a[c][h] = lo[h]; a[c][h + 8] = hi[h]; }
  }

  float f[8] = {0.f, 0.f, 0.f, 0.f, 0.f, 0.f, 0.f, 0.f};

#pragma unroll
  for (int j = 0; j < 8; ++j) {
    const int col = j * 16 + m16;   // output column (and its N-lane)
    v8f acc = {};
#pragma unroll
    for (int c = 0; c < 4; ++c) {
      // B-operand (32x16): lanes 0-15 hold K {0..7, 16..23}, lanes 16-31 K {8..15, 24..31}.
      // B[k][col] = A[k][col] = A[col][k] (symmetric) -> contiguous row read.
      const int kb = c * 32 + half * 8;
      v8h blo = *(const v8h*)&sAmat[col * DD + kb];
      v8h bhi = *(const v8h*)&sAmat[col * DD + kb + 16];
      v16h b;
#pragma unroll
      for (int h = 0; h < 8; ++h) { b[h] = blo[h]; b[h + 8] = bhi[h]; }
      acc = __builtin_amdgcn_wmma_f32_16x16x32_f16(
          /*neg_a=*/false, a[c], /*neg_b=*/false, b,
          /*c_mod=*/(short)0, acc, /*reuse_a=*/false, /*reuse_b=*/false);
    }
    // diag(H X̂ᵀ) partials: acc[r] holds H[M][col], M = r + 8*half.
    v8h xv = *(const v8h*)&sXhatT[w][col * 16 + half * 8];
#pragma unroll
    for (int r = 0; r < 8; ++r) f[r] += acc[r] * (float)xv[r];
  }

  // Reduce each f[r] over the 16 lanes of its half (N dimension).
#pragma unroll
  for (int r = 0; r < 8; ++r) {
    f[r] += __shfl_xor(f[r], 1, 32);
    f[r] += __shfl_xor(f[r], 2, 32);
    f[r] += __shfl_xor(f[r], 4, 32);
    f[r] += __shfl_xor(f[r], 8, 32);
  }

  if (m16 == 0) {
    const int rowBase = half * 8;   // lane 0 -> rows 0..7, lane 16 -> rows 8..15
#pragma unroll
    for (int r = 0; r < 8; ++r) {
      int atom = atomBase + rowBase + r;
      if (atom < N) atomicAdd(&out[atom_seg[atom]], f[r]);
    }
  }
}

// ---------------------------------------------------------------------------
extern "C" void kernel_launch(void* const* d_in, const int* in_sizes, int n_in,
                              void* d_out, int out_size, void* d_ws, size_t ws_size,
                              hipStream_t stream) {
  (void)n_in; (void)ws_size;
  const float* power_spectrum   = (const float*)d_in[0];
  const float* support_points   = (const float*)d_in[1];
  const float* weights          = (const float*)d_in[2];
  const int*   atom_segments    = (const int*)d_in[3];
  const int*   support_segments = (const int*)d_in[4];
  // d_in[5..7] are device scalars (S, T, zeta); dims derived from sizes, zeta=2.

  const int N = in_sizes[3];          // atoms
  const int M = in_sizes[4];          // support points
  const int S = out_size;             // structures (P == 1)

  float* wm = (float*)d_ws;
  size_t amat_off = (((size_t)M * sizeof(float)) + 255) & ~(size_t)255;
  _Float16* Amat = (_Float16*)((char*)d_ws + amat_off);

  hipMemsetAsync(d_out, 0, (size_t)S * sizeof(float), stream);

  build_wm_kernel<<<(M + 255) / 256, 256, 0, stream>>>(
      weights, support_segments, wm, M);

  build_amat_kernel<<<(DD / 16) * (DD / 16), 256, 0, stream>>>(
      support_points, wm, Amat, M);

  const int atomsPerBlock = 16 * WAVES;
  const int grid = (N + atomsPerBlock - 1) / atomsPerBlock;
  gap_quadform_kernel<<<grid, 32 * WAVES, 0, stream>>>(
      power_spectrum, Amat, atom_segments, (float*)d_out, N);
}
